// QuantAttentionBlock_one_59665685676673
// MI455X (gfx1250) — matrix operand order repaired and checked by reading, not compiled, not run on hardware
//
#include <hip/hip_runtime.h>
#include <hip/hip_bf16.h>

// ---------------------------------------------------------------------------
// QuantAttentionBlock for MI455X (gfx1250, wave32, WMMA 16x16x32 f16)
//   x:[8,512,32,32] -> GN(32 groups) -> qkv GEMM -> 8-head attn (T=1024,ch=64)
//   -> proj GEMM + residual
// ---------------------------------------------------------------------------

typedef __attribute__((ext_vector_type(16))) _Float16 v16h;
typedef __attribute__((ext_vector_type(8)))  float    v8f;

#define NB    8      // batch
#define NC    512    // channels
#define NT    1024   // tokens (H*W)
#define NG    32     // groups
#define NHD   8      // heads
#define CH    64     // channels per head
#define GEPS  1e-5f

static __device__ __forceinline__ v8f wmma16(v16h a, v16h b, v8f c) {
  // D = A(16x32 f16) * B(32x16 f16) + C(16x16 f32)
  return __builtin_amdgcn_wmma_f32_16x16x32_f16(false, a, false, b,
                                                (short)0, c, false, false);
}

// ---------------------------------------------------------------------------
// Kernel: f32 -> f16 convert (for weights)
// ---------------------------------------------------------------------------
__global__ void cvt_f32_to_f16(const float* __restrict__ in,
                               _Float16* __restrict__ out, int n) {
  int i = blockIdx.x * blockDim.x + threadIdx.x;
  int stride = gridDim.x * blockDim.x;
  for (; i < n; i += stride) out[i] = (_Float16)in[i];
}

// ---------------------------------------------------------------------------
// Kernel: fused GroupNorm -> f16
// One block per (batch, group): 16 channels x 1024 tokens = 16384 contiguous
// ---------------------------------------------------------------------------
__global__ __launch_bounds__(256)
void groupnorm_kernel(const float* __restrict__ x,
                      const float* __restrict__ gw,
                      const float* __restrict__ gb,
                      _Float16* __restrict__ h) {
  const int bg  = blockIdx.x;
  const int b   = bg >> 5;          // /32
  const int g   = bg & 31;
  const int tid = threadIdx.x;
  const size_t base = ((size_t)b * NC + (size_t)g * (NC / NG)) * NT;
  const int NELEM = (NC / NG) * NT; // 16384

  float s = 0.f, s2 = 0.f;
  for (int i = tid; i < NELEM; i += 256) {
    float v = x[base + i];
    s += v; s2 += v * v;
  }
  __shared__ float red[256];
  red[tid] = s; __syncthreads();
  for (int off = 128; off > 0; off >>= 1) {
    if (tid < off) red[tid] += red[tid + off];
    __syncthreads();
  }
  float S = red[0]; __syncthreads();
  red[tid] = s2; __syncthreads();
  for (int off = 128; off > 0; off >>= 1) {
    if (tid < off) red[tid] += red[tid + off];
    __syncthreads();
  }
  float S2 = red[0];

  const float inv_n = 1.0f / (float)NELEM;
  float mean = S * inv_n;
  float var  = S2 * inv_n - mean * mean;
  float rstd = rsqrtf(var + GEPS);

  for (int i = tid; i < NELEM; i += 256) {
    int c = g * (NC / NG) + (i >> 10);   // i/1024 -> local channel
    float v = (x[base + i] - mean) * rstd * gw[c] + gb[c];
    h[base + i] = (_Float16)v;
  }
}

// ---------------------------------------------------------------------------
// Kernel: batched f16 GEMM via WMMA.
//   out[b] = A[MxK] * Bm[b][KxN] (+ bias[M]) (+ resid[b][MxN])
// Block 256 threads = 8 waves; tile 128(M) x 64(N); K step 32.
// A staged row-major in LDS; B staged transposed [n][k] so both the A
// fragment (K contiguous per lane-half) and B fragment (K contiguous,
// N = lane%16) load contiguously.
// ---------------------------------------------------------------------------
template <bool OUT_F16, bool RESID>
__global__ __launch_bounds__(256)
void gemm_wmma_kernel(const _Float16* __restrict__ A,     // [M][K]
                      const _Float16* __restrict__ Bm,    // [batch][K][N]
                      const float*    __restrict__ bias,  // [M]
                      const float*    __restrict__ resid, // [batch][M][N]
                      _Float16* __restrict__ outh,        // [batch][M][N]
                      float*    __restrict__ outf,        // [batch][M][N]
                      int M, int K, int N) {
  const int batch = blockIdx.z;
  const _Float16* Bsrc = Bm + (size_t)batch * K * N;
  const size_t obase = (size_t)batch * M * N;

  __shared__ _Float16 As[128 * 32];  // [m][k]
  __shared__ _Float16 Bs[64 * 32];   // [n][k] (transposed stage)

  const int tid  = threadIdx.x;
  const int wave = tid >> 5;
  const int lane = tid & 31;
  const int hh   = lane >> 4;   // lane half (0/1)
  const int ln   = lane & 15;

  const int m0 = blockIdx.y * 128;
  const int n0 = blockIdx.x * 64;

  v8f acc[4] = {v8f{}, v8f{}, v8f{}, v8f{}};

  for (int k0 = 0; k0 < K; k0 += 32) {
    // stage A tile: 128x32 = 4096 elems, 16 per thread
#pragma unroll
    for (int i = 0; i < 16; ++i) {
      int idx = (i << 8) + tid;
      int r = idx >> 5, c = idx & 31;
      As[idx] = A[(size_t)(m0 + r) * K + (k0 + c)];
    }
    // stage B tile transposed: 64(n) x 32(k) = 2048 elems, 8 per thread
#pragma unroll
    for (int i = 0; i < 8; ++i) {
      int idx = (i << 8) + tid;
      int k = idx >> 6, n = idx & 63;
      Bs[n * 32 + k] = Bsrc[(size_t)(k0 + k) * N + (n0 + n)];
    }
    __syncthreads();

    // A fragment: rows m = wave*16 + ln; a[j] = A[m][8h+j], a[8+j] = A[m][16+8h+j]
    v16h afrag;
    {
      const _Float16* Ap = &As[(wave * 16 + ln) * 32];
#pragma unroll
      for (int j = 0; j < 8; ++j) {
        afrag[j]     = Ap[8 * hh + j];
        afrag[8 + j] = Ap[16 + 8 * hh + j];
      }
    }
#pragma unroll
    for (int nt = 0; nt < 4; ++nt) {
      // B fragment: b[j] = B[k = 8h+j][n], b[8+j] = B[k = 16+8h+j][n]  (Bs is [n][k])
      v16h bfrag;
      const _Float16* Bp = &Bs[(nt * 16 + ln) * 32];
#pragma unroll
      for (int j = 0; j < 8; ++j) { bfrag[j] = Bp[8 * hh + j]; bfrag[8 + j] = Bp[16 + 8 * hh + j]; }
      acc[nt] = wmma16(afrag, bfrag, acc[nt]);
    }
    __syncthreads();
  }

  // epilogue: C/D layout -> row = j + 8*h, col = lane%16 (+ 16*nt)
#pragma unroll
  for (int nt = 0; nt < 4; ++nt) {
#pragma unroll
    for (int j = 0; j < 8; ++j) {
      int row = m0 + wave * 16 + j + 8 * hh;
      int col = n0 + nt * 16 + ln;
      float v = acc[nt][j] + bias[row];
      size_t o = obase + (size_t)row * N + col;
      if (RESID) v += resid[o];
      if (OUT_F16) outh[o] = (_Float16)v;
      else         outf[o] = v;
    }
  }
}

// ---------------------------------------------------------------------------
// Kernel: flash attention over T=1024, ch=64, scale folded into logits (1/8).
// Grid: (16 q-tiles, 64 batch*head). Block 128 = 4 waves, 16 q-rows each.
// qkv layout: [b][1536][1024]; head hd occupies rows [hd*192, hd*192+192)
// as (q:64, k:64, v:64)  (reference reshape B,3C -> B*nh,3ch).
// ---------------------------------------------------------------------------
__global__ __launch_bounds__(128)
void attention_kernel(const _Float16* __restrict__ qkv,
                      _Float16* __restrict__ aout) {
  const int qt = blockIdx.x;            // 0..15
  const int bh = blockIdx.y;            // 0..63
  const int b  = bh >> 3;
  const int hd = bh & 7;

  const _Float16* qb = qkv + ((size_t)b * (3 * NC) + (size_t)hd * (3 * CH)) * NT;
  const _Float16* kb = qb + (size_t)CH * NT;
  const _Float16* vb = qb + (size_t)2 * CH * NT;

  __shared__ _Float16 Qs[64 * 64];      // [t][c]
  __shared__ _Float16 Ks[64 * 64];      // [s][c]
  __shared__ _Float16 Ps[4][16 * 64];   // per-wave P tile [t][s]

  const int tid  = threadIdx.x;
  const int wave = tid >> 5;
  const int lane = tid & 31;
  const int hh   = lane >> 4;
  const int ln   = lane & 15;
  const int t0   = qt * 64;

  // stage Q transposed: Qs[t][c] = q[c][t0+t]
#pragma unroll
  for (int i = 0; i < 32; ++i) {
    int idx = (i << 7) + tid;
    int c = idx >> 6, t = idx & 63;
    Qs[t * 64 + c] = qb[(size_t)c * NT + (t0 + t)];
  }
  __syncthreads();

  // Q fragments for this wave's 16 rows (K = ch = 64 -> 2 fragments)
  v16h aq[2];
  {
    const _Float16* Qp = &Qs[(wave * 16 + ln) * 64];
#pragma unroll
    for (int kk = 0; kk < 2; ++kk)
#pragma unroll
      for (int j = 0; j < 8; ++j) {
        aq[kk][j]     = Qp[kk * 32 + 8 * hh + j];
        aq[kk][8 + j] = Qp[kk * 32 + 16 + 8 * hh + j];
      }
  }

  float mrun[8], lrun[8];
#pragma unroll
  for (int j = 0; j < 8; ++j) { mrun[j] = -1e30f; lrun[j] = 0.f; }
  v8f acc[4] = {v8f{}, v8f{}, v8f{}, v8f{}};

  for (int st = 0; st < 16; ++st) {
    const int s0 = st * 64;
    __syncthreads();  // previous Ks reads complete
    // stage K transposed: Ks[s][c] = k[c][s0+s]
#pragma unroll
    for (int i = 0; i < 32; ++i) {
      int idx = (i << 7) + tid;
      int c = idx >> 6, s = idx & 63;
      Ks[s * 64 + c] = kb[(size_t)c * NT + (s0 + s)];
    }
    __syncthreads();

    // S = (Q^T K) * 1/8  : 4 column fragments of 16x16
    v8f sfr[4];
#pragma unroll
    for (int nt = 0; nt < 4; ++nt) {
      v8f sa = {};
      const _Float16* Kp = &Ks[(nt * 16 + ln) * 64];
#pragma unroll
      for (int kk = 0; kk < 2; ++kk) {
        v16h bk;
#pragma unroll
        for (int j = 0; j < 8; ++j) { bk[j] = Kp[kk * 32 + 8 * hh + j]; bk[8 + j] = Kp[kk * 32 + 16 + 8 * hh + j]; }
        sa = wmma16(aq[kk], bk, sa);
      }
      sfr[nt] = sa * 0.125f;   // 1/sqrt(ch), both q&k scales folded
    }

    // online softmax per row (row j+8h; 64 cols across 16 lanes x 4 frags)
#pragma unroll
    for (int j = 0; j < 8; ++j) {
      float mx = fmaxf(fmaxf(sfr[0][j], sfr[1][j]), fmaxf(sfr[2][j], sfr[3][j]));
#pragma unroll
      for (int off = 1; off < 16; off <<= 1)
        mx = fmaxf(mx, __shfl_xor(mx, off, 32));
      float newm = fmaxf(mrun[j], mx);
      float corr = __expf(mrun[j] - newm);
      mrun[j] = newm;
      float rs = 0.f;
#pragma unroll
      for (int nt = 0; nt < 4; ++nt) {
        float pv = __expf(sfr[nt][j] - newm);
        rs += pv;
        Ps[wave][(j + 8 * hh) * 64 + nt * 16 + ln] = (_Float16)pv;
      }
#pragma unroll
      for (int off = 1; off < 16; off <<= 1)
        rs += __shfl_xor(rs, off, 32);
      lrun[j] = lrun[j] * corr + rs;
#pragma unroll
      for (int nt = 0; nt < 4; ++nt) acc[nt][j] = acc[nt][j] * corr;
    }
    __syncthreads();  // P tile visible (C/D layout -> A layout via LDS)

    // P fragments (A-matrix over K = s local)
    v16h ap[2];
    {
      const _Float16* Pp = &Ps[wave][ln * 64];
#pragma unroll
      for (int kk = 0; kk < 2; ++kk)
#pragma unroll
        for (int j = 0; j < 8; ++j) {
          ap[kk][j]     = Pp[kk * 32 + 8 * hh + j];
          ap[kk][8 + j] = Pp[kk * 32 + 16 + 8 * hh + j];
        }
    }
    // O += P * V^T ; V fragments straight from global (contiguous in s)
#pragma unroll
    for (int nt = 0; nt < 4; ++nt) {
      const _Float16* Vp = vb + (size_t)(nt * 16 + ln) * NT + s0;
#pragma unroll
      for (int kk = 0; kk < 2; ++kk) {
        v16h bv;
#pragma unroll
        for (int j = 0; j < 8; ++j) { bv[j] = Vp[kk * 32 + 8 * hh + j]; bv[8 + j] = Vp[kk * 32 + 16 + 8 * hh + j]; }
        acc[nt] = wmma16(ap[kk], bv, acc[nt]);
      }
    }
  }

  // write a^T back as [b][C][T]: channel = hd*64 + c, token t
#pragma unroll
  for (int nt = 0; nt < 4; ++nt) {
#pragma unroll
    for (int j = 0; j < 8; ++j) {
      int t  = t0 + wave * 16 + j + 8 * hh;
      int ch = hd * CH + nt * 16 + ln;
      float l = lrun[j];
      float v = acc[nt][j] / (l > 0.f ? l : 1.f);
      aout[((size_t)b * NC + ch) * NT + t] = (_Float16)v;
    }
  }
}

// ---------------------------------------------------------------------------
// Launch
// ---------------------------------------------------------------------------
extern "C" void kernel_launch(void* const* d_in, const int* in_sizes, int n_in,
                              void* d_out, int out_size, void* d_ws, size_t ws_size,
                              hipStream_t stream) {
  const float* x      = (const float*)d_in[0];
  const float* norm_w = (const float*)d_in[1];
  const float* norm_b = (const float*)d_in[2];
  const float* qkv_w  = (const float*)d_in[3];
  const float* qkv_b  = (const float*)d_in[4];
  const float* proj_w = (const float*)d_in[5];
  const float* proj_b = (const float*)d_in[6];
  float* out = (float*)d_out;

  // workspace carve-up (f16 buffers, 256B aligned)
  char* ws = (char*)d_ws;
  size_t off = 0;
  auto alloc = [&](size_t halfs) {
    _Float16* p = (_Float16*)(ws + off);
    off += (halfs * sizeof(_Float16) + 255) & ~(size_t)255;
    return p;
  };
  _Float16* wqkv_h  = alloc((size_t)3 * NC * NC);       // 1536x512
  _Float16* wproj_h = alloc((size_t)NC * NC);           // 512x512
  _Float16* h_h     = alloc((size_t)NB * NC * NT);      // normalized x
  _Float16* qkv_h   = alloc((size_t)NB * 3 * NC * NT);  // qkv activations
  _Float16* a_h     = alloc((size_t)NB * NC * NT);      // attention out
  (void)ws_size; (void)n_in; (void)in_sizes; (void)out_size;

  // 1) weight conversion
  cvt_f32_to_f16<<<3072, 256, 0, stream>>>(qkv_w, wqkv_h, 3 * NC * NC);
  cvt_f32_to_f16<<<1024, 256, 0, stream>>>(proj_w, wproj_h, NC * NC);

  // 2) group norm -> f16
  groupnorm_kernel<<<NB * NG, 256, 0, stream>>>(x, norm_w, norm_b, h_h);

  // 3) qkv GEMM: [1536x512] x [512x1024] per batch, f16 out + bias
  gemm_wmma_kernel<true, false>
      <<<dim3(NT / 64, (3 * NC) / 128, NB), 256, 0, stream>>>(
          wqkv_h, h_h, qkv_b, nullptr, qkv_h, nullptr, 3 * NC, NC, NT);

  // 4) flash attention per (head, q-tile)
  attention_kernel<<<dim3(NT / 64, NB * NHD), 128, 0, stream>>>(qkv_h, a_h);

  // 5) proj GEMM + bias + residual -> f32 output
  gemm_wmma_kernel<false, true>
      <<<dim3(NT / 64, NC / 128, NB), 256, 0, stream>>>(
          wproj_h, a_h, proj_b, x, nullptr, out, NC, NC, NT);
}
